// GeometricSelfAttention_70196945485982
// MI455X (gfx1250) — matrix operand order repaired and checked by reading, not compiled, not run on hardware
//
#include <hip/hip_runtime.h>
#include <stddef.h>

// ---------------------------------------------------------------------------
// Types for CDNA5 WMMA (wave32): v16bf A/B fragments, v8f accumulators.
// ---------------------------------------------------------------------------
typedef __attribute__((ext_vector_type(16))) __bf16 bf16x16;
typedef __attribute__((ext_vector_type(8)))  __bf16 bf16x8;
typedef __attribute__((ext_vector_type(8)))  float  f32x8;

#define D_MODEL 512
#define NHEAD   8
#define HEAD_DIM 64
#define HIDDEN  128
#define BATCH   2
#define SLEN    1024
#define NTOK    (BATCH * SLEN)   // 2048

// ---------------------------------------------------------------------------
// fp32 -> bf16 conversion (x is consumed only by the QKV GEMMs)
// ---------------------------------------------------------------------------
__global__ __launch_bounds__(256) void cvt_f32_to_bf16(const float* __restrict__ src,
                                                       __bf16* __restrict__ dst, int n) {
  int i = blockIdx.x * 256 + threadIdx.x;
  if (i < n) dst[i] = (__bf16)src[i];
}

// ---------------------------------------------------------------------------
// Weight pre-pass: W[k][n] f32  ->  Wt[n][k] bf16  (tiled LDS transpose).
// After this, the GEMM B-fragment (lane = column n, 16 consecutive k) is two
// contiguous 16B loads — as cheap as the A fragment.
// ---------------------------------------------------------------------------
__global__ __launch_bounds__(256) void transpose_cvt_w(const float* __restrict__ W,
                                                       __bf16* __restrict__ Wt) {
  __shared__ float tile[32][33];
  const int tx = threadIdx.x & 31;
  const int ty = threadIdx.x >> 5;                 // 0..7
  const int k0 = (blockIdx.x & 15) * 32;           // 512/32 = 16 tiles per dim
  const int n0 = (blockIdx.x >> 4) * 32;
#pragma unroll
  for (int r = 0; r < 4; ++r)
    tile[ty + r * 8][tx] = W[(size_t)(k0 + ty + r * 8) * D_MODEL + n0 + tx];
  __syncthreads();
#pragma unroll
  for (int r = 0; r < 4; ++r)
    Wt[(size_t)(n0 + ty + r * 8) * D_MODEL + k0 + tx] = (__bf16)tile[tx][ty + r * 8];
}

// ---------------------------------------------------------------------------
// GEMM:  dst[M,512] = A_bf16[M,512] @ Wt_bf16^T + bias
// One wave computes a 64x16 strip (4 stacked 16x16 tiles): per K-step of 32,
// one B fragment feeds 4 v_wmma_f32_16x16x32_bf16.
// A-fragment lane layout (16-bit A 16x32): lane L -> row (L&15);
//   k = k0 + half*16 + e
// B-fragment lane layout (16-bit B 32x16): lane L -> col (L&15);
//   k = k0 + half*16 + e   (contiguous in Wt[n][k])
// C/D layout: element v -> row v + half*8, col (L&15).
// MODE: 0 = row-major bf16 store, 1 = row-major f32 store,
//       2 = transposed bf16 store dst[col][row] (packed 8 rows per 16B store)
// ---------------------------------------------------------------------------
template <int MODE>
__global__ __launch_bounds__(256) void gemm_bf16_wmma(
    const __bf16* __restrict__ A, const __bf16* __restrict__ Wt,
    const float* __restrict__ bias, void* __restrict__ dstv) {
  const int lane = threadIdx.x & 31;
  const int wave = threadIdx.x >> 5;
  const int job = blockIdx.x * 8 + wave;   // 1024 wave-jobs
  const int tn = job & 31;                 // 32 N-tiles
  const int mg = job >> 5;                 // 32 M-groups of 64 rows
  const int hv = lane >> 4;
  const int l15 = lane & 15;
  const int colB = tn * 16 + l15;
  const int row0 = mg * 64;

  f32x8 acc[4];
#pragma unroll
  for (int mt = 0; mt < 4; ++mt) { f32x8 z = {}; acc[mt] = z; }

  for (int k0 = 0; k0 < D_MODEL; k0 += 32) {
    bf16x16 bfrag;
    {
      const __bf16* wp = Wt + (size_t)colB * D_MODEL + k0 + hv * 16;
      bf16x8 r0 = *(const bf16x8*)(wp);
      bf16x8 r1 = *(const bf16x8*)(wp + 8);
#pragma unroll
      for (int e = 0; e < 8; ++e) { bfrag[e] = r0[e]; bfrag[8 + e] = r1[e]; }
    }
#pragma unroll
    for (int mt = 0; mt < 4; ++mt) {
      bf16x16 afrag;
      const __bf16* ap = A + (size_t)(row0 + mt * 16 + l15) * D_MODEL + k0 + hv * 16;
      bf16x8 r0 = *(const bf16x8*)(ap);
      bf16x8 r1 = *(const bf16x8*)(ap + 8);
#pragma unroll
      for (int e = 0; e < 8; ++e) { afrag[e] = r0[e]; afrag[8 + e] = r1[e]; }
      acc[mt] = __builtin_amdgcn_wmma_f32_16x16x32_bf16(false, afrag, false, bfrag,
                                                        (short)0, acc[mt], false, false);
    }
  }

  const float bv = bias[colB];
  if (MODE == 2) {
    // Transposed store: C/D element v is row (row0 + mt*16 + hv*8 + v), so the
    // 8 accumulator elements are 8 consecutive rows -> one packed 16B store.
#pragma unroll
    for (int mt = 0; mt < 4; ++mt) {
      bf16x8 pk;
#pragma unroll
      for (int v = 0; v < 8; ++v) pk[v] = (__bf16)(acc[mt][v] + bv);
      *(bf16x8*)((__bf16*)dstv + (size_t)colB * NTOK + row0 + mt * 16 + hv * 8) = pk;
    }
  } else {
#pragma unroll
    for (int mt = 0; mt < 4; ++mt) {
#pragma unroll
      for (int v = 0; v < 8; ++v) {
        const int row = row0 + mt * 16 + v + hv * 8;
        const float val = acc[mt][v] + bv;
        if (MODE == 1) ((float*)dstv)[(size_t)row * D_MODEL + colB] = val;
        else           ((__bf16*)dstv)[(size_t)row * D_MODEL + colB] = (__bf16)val;
      }
    }
  }
}

// ---------------------------------------------------------------------------
// Fused attention: grid = B * S/16 blocks, block = 256 threads = 8 waves.
// Wave w == head w; all heads share one 16-row q-tile so the distance-bias
// MLP is evaluated once per (q,k) pair for all 8 heads (cooperatively, LDS).
// Flash-attention over 32-key chunks; scores & P@V via WMMA bf16.
// V is consumed in transposed layout Vt[d_global][token] so every WMMA
// operand is fed by contiguous 16B loads.
// ---------------------------------------------------------------------------
__global__ __launch_bounds__(256) void attn_fused(
    const __bf16* __restrict__ Qb, const __bf16* __restrict__ Kb,
    const __bf16* __restrict__ Vt, const float* __restrict__ positions,
    const unsigned char* __restrict__ mask,
    const float* __restrict__ dw1, const float* __restrict__ db1,
    const float* __restrict__ dw2, const float* __restrict__ db2,
    __bf16* __restrict__ Ob) {
  __shared__ float s_w1[HIDDEN];
  __shared__ float s_b1[HIDDEN];
  __shared__ float s_w2[HIDDEN * NHEAD];
  __shared__ float s_b2[NHEAD];
  __shared__ float s_qpos[16][3];
  __shared__ float s_bias[16][32][NHEAD];              // per-pair bias, all heads
  __shared__ __align__(16) __bf16 s_P[NHEAD][16][32];  // probs, per head

  const int tid = threadIdx.x;
  const int lane = tid & 31;
  const int h = tid >> 5;            // head index == wave index
  const int hv = lane >> 4;
  const int l15 = lane & 15;

  const int b = blockIdx.x / (SLEN / 16);
  const int q0 = (blockIdx.x % (SLEN / 16)) * 16;

  // Stage MLP weights + q positions into LDS.
  for (int i = tid; i < HIDDEN; i += 256) { s_w1[i] = dw1[i]; s_b1[i] = db1[i]; }
  for (int i = tid; i < HIDDEN * NHEAD; i += 256) s_w2[i] = dw2[i];
  if (tid < NHEAD) s_b2[tid] = db2[tid];
  if (tid < 48) {
    int r = tid / 3, c = tid % 3;
    s_qpos[r][c] = positions[((size_t)b * SLEN + q0 + r) * 3 + c];
  }
  __syncthreads();

  // Persistent Q A-fragments (16x64 split into two K=32 chunks).
  bf16x16 qfrag[2];
  {
    const __bf16* qp = Qb + ((size_t)b * SLEN + q0 + l15) * D_MODEL + h * HEAD_DIM;
#pragma unroll
    for (int kc = 0; kc < 2; ++kc) {
      bf16x8 r0 = *(const bf16x8*)(qp + kc * 32 + hv * 16);
      bf16x8 r1 = *(const bf16x8*)(qp + kc * 32 + hv * 16 + 8);
#pragma unroll
      for (int e = 0; e < 8; ++e) { qfrag[kc][e] = r0[e]; qfrag[kc][8 + e] = r1[e]; }
    }
  }

  float m_r[8], l_r[8];
#pragma unroll
  for (int v = 0; v < 8; ++v) { m_r[v] = -1e30f; l_r[v] = 0.0f; }
  f32x8 oacc[4];
#pragma unroll
  for (int nt = 0; nt < 4; ++nt) { f32x8 z = {}; oacc[nt] = z; }

  const float scale = 0.125f;  // 1/sqrt(HEAD_DIM)

  for (int kb = 0; kb < SLEN; kb += 32) {
    __syncthreads();  // previous chunk's LDS reads complete

    // ---- cooperative distance-bias MLP: 512 pairs / 256 threads ----
#pragma unroll
    for (int pp = 0; pp < 2; ++pp) {
      const int p = tid * 2 + pp;
      const int qi = p >> 5, kk = p & 31;
      const float* kp = &positions[((size_t)b * SLEN + kb + kk) * 3];
      const float dx = s_qpos[qi][0] - kp[0];
      const float dy = s_qpos[qi][1] - kp[1];
      const float dz = s_qpos[qi][2] - kp[2];
      const float d = sqrtf(dx * dx + dy * dy + dz * dz);
      float acc[NHEAD];
#pragma unroll
      for (int hh = 0; hh < NHEAD; ++hh) acc[hh] = s_b2[hh];
#pragma unroll 4
      for (int c = 0; c < HIDDEN; ++c) {
        const float a = fmaxf(fmaf(d, s_w1[c], s_b1[c]), 0.0f);
#pragma unroll
        for (int hh = 0; hh < NHEAD; ++hh) acc[hh] = fmaf(a, s_w2[c * NHEAD + hh], acc[hh]);
      }
#pragma unroll
      for (int hh = 0; hh < NHEAD; ++hh) s_bias[qi][kk][hh] = acc[hh];
    }
    __syncthreads();  // bias ready

    // ---- scores: two 16x16 key-half tiles, each K=64 via 2 WMMA ----
    f32x8 sc[2];
#pragma unroll
    for (int kh = 0; kh < 2; ++kh) {
      f32x8 s = {};
#pragma unroll
      for (int kc = 0; kc < 2; ++kc) {
        bf16x16 kf;
        const __bf16* kp = Kb + ((size_t)b * SLEN + kb + kh * 16 + l15) * D_MODEL +
                           h * HEAD_DIM + kc * 32 + hv * 16;
        bf16x8 r0 = *(const bf16x8*)(kp);
        bf16x8 r1 = *(const bf16x8*)(kp + 8);
#pragma unroll
        for (int e = 0; e < 8; ++e) { kf[e] = r0[e]; kf[8 + e] = r1[e]; }
        s = __builtin_amdgcn_wmma_f32_16x16x32_bf16(false, qfrag[kc], false, kf,
                                                    (short)0, s, false, false);
      }
      const unsigned char mk = mask[(size_t)b * SLEN + kb + kh * 16 + l15];
#pragma unroll
      for (int v = 0; v < 8; ++v) {
        const float val = s[v] * scale + s_bias[v + hv * 8][kh * 16 + l15][h];
        s[v] = mk ? val : -1e30f;
      }
      sc[kh] = s;
    }

    // ---- online softmax: row reductions within each 16-lane half ----
    float mnew[8], alpha[8];
#pragma unroll
    for (int v = 0; v < 8; ++v) {
      float t = fmaxf(sc[0][v], sc[1][v]);
#pragma unroll
      for (int off = 1; off < 16; off <<= 1) t = fmaxf(t, __shfl_xor(t, off, 32));
      mnew[v] = fmaxf(m_r[v], t);
      alpha[v] = __expf(m_r[v] - mnew[v]);
      m_r[v] = mnew[v];
    }
#pragma unroll
    for (int v = 0; v < 8; ++v) {
      const float p0 = __expf(sc[0][v] - mnew[v]);
      const float p1 = __expf(sc[1][v] - mnew[v]);
      float t = p0 + p1;
#pragma unroll
      for (int off = 1; off < 16; off <<= 1) t += __shfl_xor(t, off, 32);
      l_r[v] = l_r[v] * alpha[v] + t;
      s_P[h][v + hv * 8][l15]      = (__bf16)p0;   // C/D layout -> LDS
      s_P[h][v + hv * 8][16 + l15] = (__bf16)p1;
    }
#pragma unroll
    for (int nt = 0; nt < 4; ++nt)
#pragma unroll
      for (int v = 0; v < 8; ++v) oacc[nt][v] *= alpha[v];
    __syncthreads();  // P tile visible across lanes

    // ---- P @ V : reload P in A layout, 4 WMMA over the 64-wide output ----
    bf16x16 pf;
    {
      const __bf16* pr = &s_P[h][l15][0];
      bf16x8 r0 = *(const bf16x8*)(pr + hv * 16);
      bf16x8 r1 = *(const bf16x8*)(pr + hv * 16 + 8);
#pragma unroll
      for (int e = 0; e < 8; ++e) { pf[e] = r0[e]; pf[8 + e] = r1[e]; }
    }
#pragma unroll
    for (int nt = 0; nt < 4; ++nt) {
      bf16x16 vf;
      // Vt[d_global][token]: lane -> d = h*64 + nt*16 + l15 (fixed),
      // element e -> key kb + hv*16 + e (contiguous 16 -> two 16B loads).
      const __bf16* vp = Vt + (size_t)(h * HEAD_DIM + nt * 16 + l15) * NTOK +
                         b * SLEN + kb + hv * 16;
      bf16x8 r0 = *(const bf16x8*)(vp);
      bf16x8 r1 = *(const bf16x8*)(vp + 8);
#pragma unroll
      for (int e = 0; e < 8; ++e) { vf[e] = r0[e]; vf[8 + e] = r1[e]; }
      oacc[nt] = __builtin_amdgcn_wmma_f32_16x16x32_bf16(false, pf, false, vf,
                                                         (short)0, oacc[nt], false, false);
    }
  }

  // ---- epilogue: normalize and emit bf16 for the output projection ----
#pragma unroll
  for (int v = 0; v < 8; ++v) {
    const float inv = 1.0f / l_r[v];
    const int q = q0 + v + hv * 8;
#pragma unroll
    for (int nt = 0; nt < 4; ++nt)
      Ob[((size_t)b * SLEN + q) * D_MODEL + h * HEAD_DIM + nt * 16 + l15] =
          (__bf16)(oacc[nt][v] * inv);
  }
}

// ---------------------------------------------------------------------------
// Host-side launcher
// ---------------------------------------------------------------------------
extern "C" void kernel_launch(void* const* d_in, const int* in_sizes, int n_in,
                              void* d_out, int out_size, void* d_ws, size_t ws_size,
                              hipStream_t stream) {
  const float* x   = (const float*)d_in[0];
  const float* pos = (const float*)d_in[1];
  const unsigned char* mask = (const unsigned char*)d_in[2];
  const float* wq = (const float*)d_in[3];
  const float* bq = (const float*)d_in[4];
  const float* wk = (const float*)d_in[5];
  const float* bk = (const float*)d_in[6];
  const float* wv = (const float*)d_in[7];
  const float* bv = (const float*)d_in[8];
  const float* dw1 = (const float*)d_in[9];
  const float* db1 = (const float*)d_in[10];
  const float* dw2 = (const float*)d_in[11];
  const float* db2 = (const float*)d_in[12];
  const float* wo = (const float*)d_in[13];
  const float* bo = (const float*)d_in[14];

  char* ws = (char*)d_ws;
  __bf16* xb  = (__bf16*)ws;                        // 2 MB (reused as O)
  __bf16* Qb  = (__bf16*)(ws + (2u << 20));         // 2 MB
  __bf16* Kb  = (__bf16*)(ws + (4u << 20));         // 2 MB
  __bf16* Vt  = (__bf16*)(ws + (6u << 20));         // 2 MB, transposed [d][token]
  __bf16* wqt = (__bf16*)(ws + (8u << 20));         // 512 KB each
  __bf16* wkt = (__bf16*)(ws + (8u << 20) + (512u << 10));
  __bf16* wvt = (__bf16*)(ws + (9u << 20));
  __bf16* wot = (__bf16*)(ws + (9u << 20) + (512u << 10));
  __bf16* Ob  = xb;                                 // x consumed before attn

  const int n = NTOK * D_MODEL;                     // 1,048,576
  cvt_f32_to_bf16<<<n / 256, 256, 0, stream>>>(x, xb, n);

  transpose_cvt_w<<<256, 256, 0, stream>>>(wq, wqt);
  transpose_cvt_w<<<256, 256, 0, stream>>>(wk, wkt);
  transpose_cvt_w<<<256, 256, 0, stream>>>(wv, wvt);
  transpose_cvt_w<<<256, 256, 0, stream>>>(wo, wot);

  const int gemm_blocks = (NTOK / 64) * (D_MODEL / 16) / 8;  // 128
  gemm_bf16_wmma<0><<<gemm_blocks, 256, 0, stream>>>(xb, wqt, bq, Qb);
  gemm_bf16_wmma<0><<<gemm_blocks, 256, 0, stream>>>(xb, wkt, bk, Kb);
  gemm_bf16_wmma<2><<<gemm_blocks, 256, 0, stream>>>(xb, wvt, bv, Vt);

  attn_fused<<<BATCH * SLEN / 16, 256, 0, stream>>>(Qb, Kb, Vt, pos, mask,
                                                    dw1, db1, dw2, db2, Ob);

  gemm_bf16_wmma<1><<<gemm_blocks, 256, 0, stream>>>(Ob, wot, bo, (float*)d_out);
}
